// VoxurfF_53841710023277
// MI455X (gfx1250) — matrix operand order repaired and checked by reading, not compiled, not run on hardware
//
#include <hip/hip_runtime.h>
#include <hip/hip_bf16.h>
#include <math.h>

// ---------------- problem constants ----------------
#define Gn 160
#define Gn2 (Gn * Gn)
#define Gn3 (Gn * Gn * Gn)
#define SAMPLES 192
#define NRAYS 2048
#define VOXEL 0.0125f
#define DISTF 0.00625f
#define NEARF 1.0f
#define THRESF 1e-4f

typedef __attribute__((ext_vector_type(16))) _Float16 v16h;
typedef __attribute__((ext_vector_type(8))) float v8f;

union ABFrag { v16h h; uint4 u[2]; };

// ---------------- workspace layout (halves for weights, then floats for bias)
// packed weight fragment buffer offsets (in _Float16 elements)
#define OW1 0        /* off  W1: NT=12 KC=4 -> 24576 */
#define OW2 24576    /* off  W2: NT=12 KC=6 -> 36864 */
#define OW3 61440    /* off  W3: NT=1  KC=6 -> 3072  */
#define EW1 64512
#define EW2 89088
#define EW3 125952
#define TW1 129024   /* tone W1: NT=8 KC=1 -> 4096 */
#define TW2 133120   /* tone W2: NT=1 KC=4 -> 2048 */
#define WPACK_HALVES 135168
#define BIAS_BYTE_OFF (WPACK_HALVES * 2)   /* 270336 */
// bias float offsets
#define OB1 0
#define OB2 192
#define OB3 384
#define EB1 400
#define EB2 592
#define EB3 784
#define TB1 800
#define TB2 928

// ---------------- LDS layout (bytes) ----------------
#define L_X      0          /* 192x128 f16  = 49152 */
#define L_H1     49152      /* 192x192 f16  = 73728 */
#define L_H2     122880     /* 192x192 f16  = 73728 */
#define L_RGBO   196608     /* 192x16  f32  = 12288 */
#define L_RGBE   208896     /* 192x16  f32  = 12288 */
#define L_EMOC   221184     /* 192x6   f32  = 4608  */
#define L_ALPHA  225792     /* 192 f32 */
#define L_W      226560     /* 192 f32 */
#define L_LIN    227328     /* 192x3 f32 = 2304 */
#define L_ACC    229632     /* 8 f32 */
#define SMEM_BYTES 229664

// ============================================================
// Weight repack: B-matrix fragment layout for v_wmma_f32_16x16x32_f16.
// dst[((nt*KC + kc)*32 + lane)*16 + h] = W[K*Nvalid + col]
//   col = nt*16 + (lane&15);  K = kc*32 + ((lane>>4)<<4) + h
// ============================================================
__global__ void pack_w_kernel(const float* __restrict__ W, _Float16* __restrict__ dst,
                              int Kvalid, int Nvalid, int KC, int NT) {
  int idx = blockIdx.x * 256 + threadIdx.x;
  int total = NT * KC * 512;
  if (idx >= total) return;
  int h    = idx & 15;
  int lane = (idx >> 4) & 31;
  int kc   = (idx >> 9) % KC;
  int nt   = idx / (KC * 512);
  int col  = nt * 16 + (lane & 15);
  int K    = kc * 32 + ((lane >> 4) << 4) + h;
  float v = (K < Kvalid && col < Nvalid) ? W[K * Nvalid + col] : 0.f;
  dst[idx] = (_Float16)v;
}

__global__ void pack_b_kernel(const float* __restrict__ b, float* __restrict__ dst,
                              int Nvalid, int Npad) {
  int i = threadIdx.x;
  if (i < Npad) dst[i] = (i < Nvalid) ? b[i] : 0.f;
}

// ============================================================
// trilinear sampling helpers
// ============================================================
struct TriW { int idx; float fx, fy, fz; };

__device__ __forceinline__ TriW make_triw(float px, float py, float pz) {
  const float scl = 0.5f * (float)(Gn - 1);
  float ux = (px + 1.f) * scl, uy = (py + 1.f) * scl, uz = (pz + 1.f) * scl;
  const float hi = (float)(Gn - 1) - 1.0e-6f;
  ux = fminf(fmaxf(ux, 0.f), hi);
  uy = fminf(fmaxf(uy, 0.f), hi);
  uz = fminf(fmaxf(uz, 0.f), hi);
  int ix = min((int)floorf(ux), Gn - 2);
  int iy = min((int)floorf(uy), Gn - 2);
  int iz = min((int)floorf(uz), Gn - 2);
  TriW w;
  w.idx = (ix * Gn + iy) * Gn + iz;
  w.fx = ux - (float)ix; w.fy = uy - (float)iy; w.fz = uz - (float)iz;
  return w;
}

__device__ __forceinline__ float tri_sample(const float* __restrict__ g, const TriW& w) {
  const float* b = g + w.idx;
  float c000 = b[0],        c001 = b[1];
  float c010 = b[Gn],       c011 = b[Gn + 1];
  float c100 = b[Gn2],      c101 = b[Gn2 + 1];
  float c110 = b[Gn2 + Gn], c111 = b[Gn2 + Gn + 1];
  float c00 = c000 + w.fx * (c100 - c000);
  float c01 = c001 + w.fx * (c101 - c001);
  float c10 = c010 + w.fx * (c110 - c010);
  float c11 = c011 + w.fx * (c111 - c011);
  float c0 = c00 + w.fy * (c10 - c00);
  float c1 = c01 + w.fy * (c11 - c01);
  return c0 + w.fz * (c1 - c0);
}

__device__ __forceinline__ float sdf_at(const float* __restrict__ g,
                                        float x, float y, float z) {
  TriW w = make_triw(x, y, z);
  return tri_sample(g, w);
}

// ============================================================
// Tiled WMMA GEMM over LDS activations + packed global weights.
// Out = act( X[M x Kpad] @ W[Kpad x N] + bias )
// mode: 0 = relu -> f16 LDS,  1 = linear -> f32 LDS,  2 = sigmoid -> f32 LDS
// ============================================================
__device__ __forceinline__ void gemm_wmma(const _Float16* __restrict__ Xlds, int xstride, int KC,
                                          const _Float16* __restrict__ Wp,
                                          const float* __restrict__ bias,
                                          int MT, int NT,
                                          _Float16* __restrict__ outH, int ostride,
                                          float* __restrict__ outF, int ofstride,
                                          int mode) {
  const int lane = threadIdx.x & 31;
  const int wave = threadIdx.x >> 5;
  const int nwaves = blockDim.x >> 5;
  const int l15 = lane & 15;
  const int lhi = lane >> 4;
  const int ntiles = MT * NT;
  for (int tile = wave; tile < ntiles; tile += nwaves) {
    int mt = tile / NT, nt = tile % NT;
    v8f acc = {0.f, 0.f, 0.f, 0.f, 0.f, 0.f, 0.f, 0.f};
    const int arow = mt * 16 + l15;
    for (int kc = 0; kc < KC; ++kc) {
      int kbase = kc * 32 + (lhi << 4);
      ABFrag a, b;
      const _Float16* ap = Xlds + arow * xstride + kbase;
      a.u[0] = *(const uint4*)(ap);
      a.u[1] = *(const uint4*)(ap + 8);
      const _Float16* bp = Wp + (((nt * KC + kc) * 32 + lane) << 4);
      b.u[0] = *(const uint4*)(bp);
      b.u[1] = *(const uint4*)(bp + 8);
      acc = __builtin_amdgcn_wmma_f32_16x16x32_f16(
          false, a.h, false, b.h, (short)0, acc, false, false);
    }
    int col = nt * 16 + l15;
    float bv = bias[col];
#pragma unroll
    for (int r = 0; r < 8; ++r) {
      int rrow = mt * 16 + r + (lhi << 3);
      float v = acc[r] + bv;
      if (mode == 0) {
        v = fmaxf(v, 0.f);
        outH[rrow * ostride + col] = (_Float16)v;
      } else if (mode == 1) {
        outF[rrow * ofstride + col] = v;
      } else {
        outF[rrow * ofstride + col] = 1.f / (1.f + expf(-v));
      }
    }
  }
}

// ============================================================
// Main fused kernel: one workgroup per ray.
// ============================================================
__global__ void __launch_bounds__(256, 1)
voxurf_main(const float* __restrict__ rays_o, const float* __restrict__ rays_d,
            const float* __restrict__ viewdirs, const int* __restrict__ em_modes,
            const int* __restrict__ s_val,
            const float* __restrict__ sdf_grid,
            const float* __restrict__ off_cg, const float* __restrict__ emo_cg,
            const _Float16* __restrict__ wp, const float* __restrict__ bp,
            float* __restrict__ out) {
  extern __shared__ char smem[];
  _Float16* X   = (_Float16*)(smem + L_X);
  _Float16* H1  = (_Float16*)(smem + L_H1);
  _Float16* H2  = (_Float16*)(smem + L_H2);
  float* rgbOff = (float*)(smem + L_RGBO);
  float* rgbEmo = (float*)(smem + L_RGBE);
  float* emoC   = (float*)(smem + L_EMOC);
  float* alphaL = (float*)(smem + L_ALPHA);
  float* wL     = (float*)(smem + L_W);
  float* linL   = (float*)(smem + L_LIN);
  float* accL   = (float*)(smem + L_ACC);

  const int ray = blockIdx.x;
  const int tid = threadIdx.x;
  const float s = (float)(*s_val);
  const int on = (em_modes[ray] == 1);
  const float rox = rays_o[ray * 3], roy = rays_o[ray * 3 + 1], roz = rays_o[ray * 3 + 2];
  const float rdx = rays_d[ray * 3], rdy = rays_d[ray * 3 + 1], rdz = rays_d[ray * 3 + 2];
  const float vdx = viewdirs[ray * 3], vdy = viewdirs[ray * 3 + 1], vdz = viewdirs[ray * 3 + 2];

  // ---------- Phase A: features ----------
  if (tid < SAMPLES) {
    float tval = NEARF + ((float)tid + 0.5f) * DISTF;
    float px = rox + tval * rdx, py = roy + tval * rdy, pz = roz + tval * rdz;
    bool inside = (px > -1.f) && (px < 1.f) && (py > -1.f) && (py < 1.f) &&
                  (pz > -1.f) && (pz < 1.f);
    TriW wc = make_triw(px, py, pz);
    float sdfv = tri_sample(sdf_grid, wc);
    _Float16* xr = X + tid * 128;

    float g0x = 0.f, g0y = 0.f, g0z = 0.f;
#pragma unroll
    for (int d = 0; d < 4; ++d) {
      float h = (float)(d + 1) * VOXEL;
      float sxm = sdf_at(sdf_grid, px - h, py, pz);
      float sxp = sdf_at(sdf_grid, px + h, py, pz);
      float sym = sdf_at(sdf_grid, px, py - h, pz);
      float syp = sdf_at(sdf_grid, px, py + h, pz);
      float szm = sdf_at(sdf_grid, px, py, pz - h);
      float szp = sdf_at(sdf_grid, px, py, pz + h);
      xr[76 + d * 6 + 0] = (_Float16)sxm;
      xr[76 + d * 6 + 1] = (_Float16)sxp;
      xr[76 + d * 6 + 2] = (_Float16)sym;
      xr[76 + d * 6 + 3] = (_Float16)syp;
      xr[76 + d * 6 + 4] = (_Float16)szm;
      xr[76 + d * 6 + 5] = (_Float16)szp;
      float inv2h = 0.5f / h;
      float gx = (sxp - sxm) * inv2h;
      float gy = (syp - sym) * inv2h;
      float gz = (szp - szm) * inv2h;
      if (d == 0) { g0x = gx; g0y = gy; g0z = gz; }
      float inv = 1.f / (sqrtf(gx * gx + gy * gy + gz * gz) + 1e-9f);
      xr[100 + d * 3 + 0] = (_Float16)(gx * inv);    // FLIP = (1,-1,-1)
      xr[100 + d * 3 + 1] = (_Float16)(-gy * inv);
      xr[100 + d * 3 + 2] = (_Float16)(-gz * inv);
    }
    // alpha
    float true_cos = vdx * g0x + vdy * g0y + vdz * g0z;
    float iter_cos = fminf(true_cos, 0.f);
    float pc = 1.f / (1.f + expf(-(sdfv - iter_cos * DISTF * 0.5f) * s));
    float nc = 1.f / (1.f + expf(-(sdfv + iter_cos * DISTF * 0.5f) * s));
    float alpha = (pc - nc + 1e-5f) / (pc + 1e-5f);
    alpha = fminf(fmaxf(alpha, 0.f), 1.f);
    alphaL[tid] = (inside && alpha > THRESF) ? alpha : 0.f;

    // colors (shared corner weights)
#pragma unroll
    for (int c = 0; c < 6; ++c) {
      xr[c] = (_Float16)tri_sample(off_cg + c * Gn3, wc);
      emoC[tid * 6 + c] = tri_sample(emo_cg + c * Gn3, wc);
    }
    // xyz + positional encodings
    float xyz[3] = {(px + 1.f) * 0.5f, (py + 1.f) * 0.5f, (pz + 1.f) * 0.5f};
    xr[6] = (_Float16)xyz[0]; xr[7] = (_Float16)xyz[1]; xr[8] = (_Float16)xyz[2];
#pragma unroll
    for (int k = 0; k < 3; ++k) {
#pragma unroll
      for (int j = 0; j < 5; ++j) {
        float e = xyz[k] * (float)(1 << j);
        xr[9 + k * 5 + j]  = (_Float16)sinf(e);
        xr[24 + k * 5 + j] = (_Float16)cosf(e);
      }
    }
    float vdv[3] = {vdx, vdy, vdz};
#pragma unroll
    for (int k = 0; k < 3; ++k) {
#pragma unroll
      for (int j = 0; j < 4; ++j) {
        float e = vdv[k] * (float)(1 << j);
        xr[39 + k * 4 + j] = (_Float16)e;
        xr[51 + k * 4 + j] = (_Float16)sinf(e);
        xr[63 + k * 4 + j] = (_Float16)cosf(e);
      }
    }
    xr[75] = (_Float16)sdfv;
#pragma unroll
    for (int k = 112; k < 128; ++k) xr[k] = (_Float16)0.f;
  }
  if (tid < 8) accL[tid] = 0.f;
  __syncthreads();

  // ---------- Phase B: cumprod weights ----------
  if (tid == 0) {
    float T = 1.f;
    for (int i = 0; i < SAMPLES; ++i) {
      float a = alphaL[i];
      float wv = a * T;
      T *= (1.f - a);
      wL[i] = (wv > THRESF) ? wv : 0.f;
    }
    out[6 * NRAYS + ray] = T;  // alphainv_last
  }
  __syncthreads();

  // ---------- Phase C: off branch ----------
  gemm_wmma(X, 128, 4, wp + OW1, bp + OB1, 12, 12, H1, 192, nullptr, 0, 0);
  __syncthreads();
  gemm_wmma(H1, 192, 6, wp + OW2, bp + OB2, 12, 12, H2, 192, nullptr, 0, 0);
  __syncthreads();
  gemm_wmma(H2, 192, 6, wp + OW3, bp + OB3, 12, 1, nullptr, 0, rgbOff, 16, 1);
  __syncthreads();

  // ---------- Phase D: emo branch (patch color columns) ----------
  if (tid < SAMPLES) {
#pragma unroll
    for (int c = 0; c < 6; ++c) X[tid * 128 + c] = (_Float16)emoC[tid * 6 + c];
  }
  __syncthreads();
  gemm_wmma(X, 128, 4, wp + EW1, bp + EB1, 12, 12, H1, 192, nullptr, 0, 0);
  __syncthreads();
  gemm_wmma(H1, 192, 6, wp + EW2, bp + EB2, 12, 12, H2, 192, nullptr, 0, 0);
  __syncthreads();
  gemm_wmma(H2, 192, 6, wp + EW3, bp + EB3, 12, 1, nullptr, 0, rgbEmo, 16, 1);
  __syncthreads();

  // ---------- Phase E: lin_rgb + tonemap input ----------
  if (tid < SAMPLES) {
    float lin[3];
#pragma unroll
    for (int c = 0; c < 3; ++c) {
      float o = rgbOff[tid * 16 + c];
      float e = rgbEmo[tid * 16 + c];
      float l = on ? (e + o) : o;
      lin[c] = l;
      linL[tid * 3 + c] = l;
    }
    _Float16* xr = X + tid * 32;  // reuse X as [192][32]
    xr[0] = (_Float16)lin[0]; xr[1] = (_Float16)lin[1]; xr[2] = (_Float16)lin[2];
#pragma unroll
    for (int c = 0; c < 3; ++c) {
#pragma unroll
      for (int j = 0; j < 4; ++j) {
        float e = lin[c] * (float)(1 << j);
        xr[3 + c * 4 + j]  = (_Float16)sinf(e);
        xr[15 + c * 4 + j] = (_Float16)cosf(e);
      }
    }
#pragma unroll
    for (int k = 27; k < 32; ++k) xr[k] = (_Float16)0.f;
  }
  __syncthreads();
  gemm_wmma(X, 32, 1, wp + TW1, bp + TB1, 12, 8, H1, 128, nullptr, 0, 0);
  __syncthreads();
  gemm_wmma(H1, 128, 4, wp + TW2, bp + TB2, 12, 1, nullptr, 0, rgbOff, 16, 2);
  __syncthreads();

  // ---------- Phase F: composite ----------
  if (tid < SAMPLES) {
    float wv = wL[tid];
    if (wv != 0.f) {
#pragma unroll
      for (int c = 0; c < 3; ++c) {
        atomicAdd(&accL[c], wv * rgbOff[tid * 16 + c]);
        atomicAdd(&accL[3 + c], wv * linL[tid * 3 + c]);
      }
    }
  }
  __syncthreads();
  if (tid < 3) {
    out[ray * 3 + tid] = accL[tid];
    out[3 * NRAYS + ray * 3 + tid] = accL[3 + tid];
  }
}

// ============================================================
extern "C" void kernel_launch(void* const* d_in, const int* in_sizes, int n_in,
                              void* d_out, int out_size, void* d_ws, size_t ws_size,
                              hipStream_t stream) {
  const float* rays_o   = (const float*)d_in[0];
  const float* rays_d   = (const float*)d_in[1];
  const float* viewdirs = (const float*)d_in[2];
  const int*   em_modes = (const int*)d_in[3];
  const int*   s_val    = (const int*)d_in[4];
  const float* sdf_grid = (const float*)d_in[5];
  const float* off_cg   = (const float*)d_in[6];
  const float* emo_cg   = (const float*)d_in[7];
  // off_params: (W 112x192, b192), (W 192x192, b192), (W 192x3, b3)   idx 8..13
  // emo_params: idx 14..19
  // tone_params: (W 27x128, b128), (W 128x3, b3)                      idx 20..23

  _Float16* wpack = (_Float16*)d_ws;
  float* bpack = (float*)((char*)d_ws + BIAS_BYTE_OFF);

  auto packW = [&](int i, int off, int Kv, int Nv, int KC, int NT) {
    int total = NT * KC * 512;
    pack_w_kernel<<<(total + 255) / 256, 256, 0, stream>>>(
        (const float*)d_in[i], wpack + off, Kv, Nv, KC, NT);
  };
  auto packB = [&](int i, int off, int Nv, int Npad) {
    pack_b_kernel<<<1, 256, 0, stream>>>((const float*)d_in[i], bpack + off, Nv, Npad);
  };

  packW(8,  OW1, 112, 192, 4, 12);
  packW(10, OW2, 192, 192, 6, 12);
  packW(12, OW3, 192, 3,   6, 1);
  packW(14, EW1, 112, 192, 4, 12);
  packW(16, EW2, 192, 192, 6, 12);
  packW(18, EW3, 192, 3,   6, 1);
  packW(20, TW1, 27,  128, 1, 8);
  packW(22, TW2, 128, 3,   4, 1);

  packB(9,  OB1, 192, 192);
  packB(11, OB2, 192, 192);
  packB(13, OB3, 3,   16);
  packB(15, EB1, 192, 192);
  packB(17, EB2, 192, 192);
  packB(19, EB3, 3,   16);
  packB(21, TB1, 128, 128);
  packB(23, TB2, 3,   16);

  voxurf_main<<<NRAYS, 256, SMEM_BYTES, stream>>>(
      rays_o, rays_d, viewdirs, em_modes, s_val,
      sdf_grid, off_cg, emo_cg, wpack, bpack, (float*)d_out);
}
